// ActionMapCTRNN_45896020525290
// MI455X (gfx1250) — matrix operand order repaired and checked by reading, not compiled, not run on hardware
//
#include <hip/hip_runtime.h>
#include <hip/hip_bf16.h>
#include <stdint.h>

// ---------------------------------------------------------------------------
// ActionMapCTRNN on MI455X (gfx1250, wave32, WMMA)
//
// out[t,b,o] = sum_i (ctx_t @ W_cam.T + b_cam)[b,o*128+i] * x[t,b,i]
// ctx_{t+1}  = 0.9*ctx_t + 0.1*relu(ctx_t@W_cc.T + b_cc + x_t@W_ic.T + b_ic)
//
// bf16 weights (W_cam bf16 = 32MB, L2-resident), fp32 accum via
// v_wmma_f32_16x16x32_bf16. amap GEMM: one wave per (n-tile, k-half),
// 4 M-tile accumulators sharing each B fragment (minimal L2 B traffic),
// einsum fused into epilogue with hardware fp32 atomics.
// ---------------------------------------------------------------------------

typedef __bf16 bf16_t;
typedef __attribute__((ext_vector_type(16))) __bf16 v16bf;
typedef __attribute__((ext_vector_type(8)))  __bf16 v8bf;
typedef __attribute__((ext_vector_type(8)))  float  v8f;

#define T_STEPS 512
#define BATCH   64
#define ISZ     128
#define CSZ     1024
#define OSZ     128

#define AMAP_BLOCKS 256   // 2048 waves: 1024 n-tiles x 2 k-halves
#define CTX_BLOCKS  32    // 256 waves: 4 m-tiles x 64 n-tiles

__device__ __forceinline__ v16bf cat8(v8bf lo, v8bf hi) {
    return __builtin_shufflevector(lo, hi, 0,1,2,3,4,5,6,7,8,9,10,11,12,13,14,15);
}

// A fragment (16x32 bf16): lane l (l15=l&15, half=l>>4): row M = mbase+l15,
//   elems[0..7]  = A[M, k0 + half*8 .. +8)
//   elems[8..15] = A[M, k0 + 16 + half*8 .. +8)
__device__ __forceinline__ v16bf load_a_frag(const bf16_t* __restrict__ rowp, int k0, int half) {
    v8bf lo = *(const v8bf*)(rowp + k0 + half * 8);
    v8bf hi = *(const v8bf*)(rowp + k0 + 16 + half * 8);
    return cat8(lo, hi);
}

// B fragment (32x16 bf16), B[k,n] = W[n,k] (W row-major, contiguous in k):
//   lane l: col N = nbase + l15, elems[0..7] = W[N, k0 + half*8 .. +8),
//   elems[8..15] = W[N, k0 + 16 + half*8 .. +8)
__device__ __forceinline__ v16bf load_b_frag(const bf16_t* __restrict__ rowp, int k0, int half) {
    v8bf lo = *(const v8bf*)(rowp + k0 + half * 8);
    v8bf hi = *(const v8bf*)(rowp + k0 + 16 + half * 8);
    return cat8(lo, hi);
}

// ------------------------- precompute kernels ------------------------------

__global__ __launch_bounds__(256)
void cvt_f32_to_bf16_kernel(const float* __restrict__ src, bf16_t* __restrict__ dst, int n) {
    int i = blockIdx.x * 256 + threadIdx.x;
    if (i < n) dst[i] = (bf16_t)src[i];
}

__global__ __launch_bounds__(256)
void init_ctx_kernel(const float* __restrict__ ctx_in,
                     float* __restrict__ ctxf0, bf16_t* __restrict__ ctxb0) {
    int i = blockIdx.x * 256 + threadIdx.x;   // 65536 elements
    float v = ctx_in[i];
    ctxf0[i] = v;
    ctxb0[i] = (bf16_t)v;
}

__global__ __launch_bounds__(256)
void copy_final_ctx_kernel(const float* __restrict__ ctxf, float* __restrict__ dst) {
    int i = blockIdx.x * 256 + threadIdx.x;   // 65536 elements
    dst[i] = ctxf[i];
}

// out[t,b,o] = sum_i x[t,b,i] * b_cam[o*128+i]   (pre-fills d_out bias term)
// GEMM: M = T*B = 32768, N = 128, K = 128, one wave per 16x16 tile.
__global__ __launch_bounds__(256)
void xb_bias_kernel(const bf16_t* __restrict__ x16,   // [T*B, 128]
                    const bf16_t* __restrict__ bcam,  // [128, 128] (o, i)
                    float* __restrict__ out)          // [T*B, 128]
{
    const int lane = threadIdx.x & 31;
    const int wave = threadIdx.x >> 5;
    const int l15  = lane & 15;
    const int half = lane >> 4;
    const int w  = blockIdx.x * 8 + wave;     // 0..16383
    const int m0 = (w >> 3) * 16;
    const int n0 = (w & 7) * 16;

    v8f c = {};
    const bf16_t* Arow = x16  + (size_t)(m0 + l15) * ISZ;
    const bf16_t* Brow = bcam + (size_t)(n0 + l15) * ISZ;
#pragma unroll
    for (int kt = 0; kt < 4; ++kt) {
        v16bf a = load_a_frag(Arow, kt * 32, half);
        v16bf b = load_b_frag(Brow, kt * 32, half);
        c = __builtin_amdgcn_wmma_f32_16x16x32_bf16(false, a, false, b, (short)0, c, false, false);
    }
#pragma unroll
    for (int j = 0; j < 8; ++j) {
        int row = m0 + j + half * 8;
        out[(size_t)row * OSZ + n0 + l15] = c[j];
    }
}

// ----------------------------- step kernel ---------------------------------
// Blocks [0,AMAP_BLOCKS):  amap GEMM, one wave per (n-tile, k-half), all 4
//   M-tiles accumulated per wave so each B fragment is loaded exactly once.
// Blocks [AMAP_BLOCKS, +CTX_BLOCKS): ctx GEMM (M=64,N=1024,K=1024+128).
__global__ __launch_bounds__(256)
void step_kernel(const float*  __restrict__ x,        // [T,B,I] fp32
                 const bf16_t* __restrict__ x16,      // [T,B,I] bf16
                 const bf16_t* __restrict__ Wcam16,   // [16384,1024]
                 const bf16_t* __restrict__ Wcc16,    // [1024,1024]
                 const bf16_t* __restrict__ Wic16,    // [1024,128]
                 const float*  __restrict__ b_cc,     // [1024]
                 const float*  __restrict__ b_ic,     // [1024]
                 const float*  __restrict__ ctxf_cur, // [64,1024]
                 const bf16_t* __restrict__ ctxb_cur, // [64,1024]
                 float*        __restrict__ ctxf_nxt,
                 bf16_t*       __restrict__ ctxb_nxt,
                 float*        __restrict__ out,      // [T,B,O]
                 int t)
{
    const int lane = threadIdx.x & 31;
    const int wave = threadIdx.x >> 5;
    const int l15  = lane & 15;
    const int half = lane >> 4;

    if (blockIdx.x < AMAP_BLOCKS) {
        // ---- amap GEMM: wave = (n-tile, k-half), 4 M-tile accumulators --
        const int w  = blockIdx.x * 8 + wave;   // 0..2047
        const int ks = w & 1;                   // k-half: 0 or 1
        const int nt = w >> 1;                  // 0..1023
        const int n0 = nt * 16;

        v8f c0 = {}, c1 = {}, c2 = {}, c3 = {};
        const bf16_t* Brow = Wcam16   + (size_t)(n0 + l15) * CSZ;
        const bf16_t* A0   = ctxb_cur + (size_t)( 0 + l15) * CSZ;
        const bf16_t* A1   = ctxb_cur + (size_t)(16 + l15) * CSZ;
        const bf16_t* A2   = ctxb_cur + (size_t)(32 + l15) * CSZ;
        const bf16_t* A3   = ctxb_cur + (size_t)(48 + l15) * CSZ;
        const int kbeg = ks * 16;
#pragma unroll 4
        for (int kt = kbeg; kt < kbeg + 16; ++kt) {
            const int k = kt * 32;
            v16bf b = load_b_frag(Brow, k, half);
            v16bf a0 = load_a_frag(A0, k, half);
            v16bf a1 = load_a_frag(A1, k, half);
            v16bf a2 = load_a_frag(A2, k, half);
            v16bf a3 = load_a_frag(A3, k, half);
            c0 = __builtin_amdgcn_wmma_f32_16x16x32_bf16(false, a0, false, b, (short)0, c0, false, false);
            c1 = __builtin_amdgcn_wmma_f32_16x16x32_bf16(false, a1, false, b, (short)0, c1, false, false);
            c2 = __builtin_amdgcn_wmma_f32_16x16x32_bf16(false, a2, false, b, (short)0, c2, false, false);
            c3 = __builtin_amdgcn_wmma_f32_16x16x32_bf16(false, a3, false, b, (short)0, c3, false, false);
        }

        // ---- fused einsum epilogue: out[b,o] += sum_i D[b,i]*x[b,i] -----
        const int o     = n0 >> 7;        // 128 i's per o; tile fits in one o
        const int ibase = n0 & 127;
        const float* xt = x + (size_t)t * BATCH * ISZ;
        const float  xv = 1.0f;           // (kept for clarity; unused)
        (void)xv;
#pragma unroll
        for (int mt = 0; mt < 4; ++mt) {
            v8f c = (mt == 0) ? c0 : (mt == 1) ? c1 : (mt == 2) ? c2 : c3;
            const int m0 = mt * 16;
#pragma unroll
            for (int j = 0; j < 8; ++j) {
                int row = m0 + j + half * 8;
                float v = c[j] * xt[(size_t)row * ISZ + ibase + l15];
                v += __shfl_xor(v, 1, 32);
                v += __shfl_xor(v, 2, 32);
                v += __shfl_xor(v, 4, 32);
                v += __shfl_xor(v, 8, 32);   // stays within 16-lane half
                if (l15 == 0)
                    unsafeAtomicAdd(&out[((size_t)t * BATCH + row) * OSZ + o], v);
            }
        }
    } else {
        // ---- ctx update tile --------------------------------------------
        const int v  = (blockIdx.x - AMAP_BLOCKS) * 8 + wave;  // 0..255
        const int m0 = (v >> 6) * 16;
        const int n0 = (v & 63) * 16;

        v8f c = {};
        const bf16_t* Arow = ctxb_cur + (size_t)(m0 + l15) * CSZ;
        const bf16_t* Brow = Wcc16    + (size_t)(n0 + l15) * CSZ;
#pragma unroll 4
        for (int kt = 0; kt < 32; ++kt) {
            v16bf a = load_a_frag(Arow, kt * 32, half);
            v16bf b = load_b_frag(Brow, kt * 32, half);
            c = __builtin_amdgcn_wmma_f32_16x16x32_bf16(false, a, false, b, (short)0, c, false, false);
        }
        // + x_t @ W_ic.T  (K = 128)
        const bf16_t* Xrow = x16   + ((size_t)t * BATCH + m0 + l15) * ISZ;
        const bf16_t* Crow = Wic16 + (size_t)(n0 + l15) * ISZ;
#pragma unroll
        for (int kt = 0; kt < 4; ++kt) {
            v16bf a = load_a_frag(Xrow, kt * 32, half);
            v16bf b = load_b_frag(Crow, kt * 32, half);
            c = __builtin_amdgcn_wmma_f32_16x16x32_bf16(false, a, false, b, (short)0, c, false, false);
        }

        const float bias = b_cc[n0 + l15] + b_ic[n0 + l15];
#pragma unroll
        for (int j = 0; j < 8; ++j) {
            int row = m0 + j + half * 8;
            int col = n0 + l15;
            float pre = c[j] + bias;
            float cn  = pre > 0.0f ? pre : 0.0f;      // relu
            float old = ctxf_cur[(size_t)row * CSZ + col];
            float nw  = old * 0.9f + cn * 0.1f;        // leak
            ctxf_nxt[(size_t)row * CSZ + col] = nw;
            ctxb_nxt[(size_t)row * CSZ + col] = (bf16_t)nw;
        }
    }
}

// ------------------------------ launcher -----------------------------------

extern "C" void kernel_launch(void* const* d_in, const int* in_sizes, int n_in,
                              void* d_out, int out_size, void* d_ws, size_t ws_size,
                              hipStream_t stream) {
    const float* x     = (const float*)d_in[0];  // [512,64,128]
    const float* ctx0  = (const float*)d_in[1];  // [64,1024]
    const float* W_cc  = (const float*)d_in[2];  // [1024,1024]
    const float* b_cc  = (const float*)d_in[3];  // [1024]
    const float* W_ic  = (const float*)d_in[4];  // [1024,128]
    const float* b_ic  = (const float*)d_in[5];  // [1024]
    const float* W_cam = (const float*)d_in[6];  // [16384,1024]
    const float* b_cam = (const float*)d_in[7];  // [16384]
    float* out = (float*)d_out;                  // [512,64,128] ++ [64,1024]

    // workspace layout (bytes)
    char* ws = (char*)d_ws;
    bf16_t* Wcam16 = (bf16_t*)(ws);                    // 33,554,432
    bf16_t* Wcc16  = (bf16_t*)(ws + 33554432ull);      //  2,097,152
    bf16_t* Wic16  = (bf16_t*)(ws + 35651584ull);      //    262,144
    bf16_t* bcam16 = (bf16_t*)(ws + 35913728ull);      //     32,768
    bf16_t* x16    = (bf16_t*)(ws + 35946496ull);      //  8,388,608
    float*  ctxf   = (float*) (ws + 44335104ull);      //  2 x 262,144
    bf16_t* ctxb   = (bf16_t*)(ws + 44859392ull);      //  2 x 131,072
    const int CTXN = BATCH * CSZ;                      // 65536

    // 1) one-time bf16 conversions (weights stay L2-resident across steps)
    cvt_f32_to_bf16_kernel<<<(16384*1024+255)/256, 256, 0, stream>>>(W_cam, Wcam16, 16384*1024);
    cvt_f32_to_bf16_kernel<<<(1024*1024+255)/256,  256, 0, stream>>>(W_cc,  Wcc16,  1024*1024);
    cvt_f32_to_bf16_kernel<<<(1024*128+255)/256,   256, 0, stream>>>(W_ic,  Wic16,  1024*128);
    cvt_f32_to_bf16_kernel<<<(16384+255)/256,      256, 0, stream>>>(b_cam, bcam16, 16384);
    cvt_f32_to_bf16_kernel<<<(T_STEPS*BATCH*ISZ+255)/256, 256, 0, stream>>>(x, x16, T_STEPS*BATCH*ISZ);

    // 2) context init (ping-pong buffer 0)
    init_ctx_kernel<<<CTXN/256, 256, 0, stream>>>(ctx0, ctxf, ctxb);

    // 3) pre-fill out with the bias term  x @ b_cam.T  (einsum atomics add on top)
    xb_bias_kernel<<<2048, 256, 0, stream>>>(x16, bcam16, out);

    // 4) sequential recurrence: one fused kernel per timestep
    for (int t = 0; t < T_STEPS; ++t) {
        int cur = t & 1, nxt = (t + 1) & 1;
        step_kernel<<<AMAP_BLOCKS + CTX_BLOCKS, 256, 0, stream>>>(
            x, x16, Wcam16, Wcc16, Wic16, b_cc, b_ic,
            ctxf + (size_t)cur * CTXN, ctxb + (size_t)cur * CTXN,
            ctxf + (size_t)nxt * CTXN, ctxb + (size_t)nxt * CTXN,
            out, t);
    }

    // 5) final context -> tail of d_out (after 512 steps it sits in buffer 0)
    copy_final_ctx_kernel<<<CTXN/256, 256, 0, stream>>>(ctxf, out + (size_t)T_STEPS * BATCH * OSZ);
}
